// X3SelfAttention_35227321762418
// MI455X (gfx1250) — compile-verified, matrix operand order checked by reading, not run on hardware
//
#include <hip/hip_runtime.h>
#include <math.h>
#include <stdint.h>

// ---------------------------------------------------------------------------
// X3SelfAttention for MI455X (gfx1250, wave32, WMMA + TDM).
//
// Roofline: GEMM hidden@Wh^T = 137 GFLOP; hidden = 512 MiB f32 streams from
// HBM twice (score pass + weighted-sum pass) => ~1.07 GB => ~46 us at
// 23.3 TB/s; WMMA compute is <1 us => purely memory bound. So: fuse the
// score computation (GEMM + bias + relu + V-dot) into one streaming pass with
// v_wmma_f32_16x16x32_bf16, stage hidden tiles into LDS with the Tensor Data
// Mover (tensor_load_to_lds / s_wait_tensorcnt), and never materialize the
// 256 MiB Ah tensor.
// ---------------------------------------------------------------------------

#define BB 64
#define SS 2048
#define HH 1024
#define AA 512
#define XX0 1024
#define XX1 512
#define XX2 512

typedef __attribute__((ext_vector_type(16))) __bf16        v16bf;
typedef __attribute__((ext_vector_type(8)))  __bf16        v8bf;
typedef __attribute__((ext_vector_type(8)))  float         v8f;
typedef __attribute__((ext_vector_type(4)))  unsigned int  v4u;
typedef __attribute__((ext_vector_type(8)))  int           v8i;
typedef __attribute__((ext_vector_type(4)))  int           v4i;

#define HPAD 1032   // 1024 + 8 bf16 pad: row stride 2064 B => 4-bank step, no conflicts

// TDM availability: device pass has the builtin; host pass / older toolchains
// fall back to manual staging (still correct, still WMMA).
#if defined(__HIP_DEVICE_COMPILE__) && __has_builtin(__builtin_amdgcn_tensor_load_to_lds)
#define HAVE_TDM 1
#else
#define HAVE_TDM 0
#endif

// ---------------------------------------------------------------------------
// Kernel 0: repack Wh_w (f32 [A,H] row-major) into bf16 tiles pre-swizzled in
// the WMMA B-fragment register order (ISA 7.12.2, 16-bit operands):
//   lane = n + 16*hi ; elem e: j=e/2, p=e&1,
//   k = 2*j + p + (hi ? (j<4 ? 8 : 16) : (j<4 ? 0 : 8))
// Tile (t, kc) covers a = t*16+n (N) and h = kc*32+k (K).
// Layout: wswz[(t*32+kc)*512 + lane*16 + e]
// ---------------------------------------------------------------------------
__global__ void k_swizzle_wh(const float* __restrict__ Wh, __bf16* __restrict__ wswz) {
    int i    = blockIdx.x * blockDim.x + threadIdx.x;   // 32*32*512 total
    int tile = i >> 9;
    int t    = tile >> 5;
    int kc   = tile & 31;
    int idx  = i & 511;
    int lane = idx >> 4;
    int e    = idx & 15;
    int n    = lane & 15;
    int hi   = lane >> 4;
    int j    = e >> 1;
    int p    = e & 1;
    int base = hi ? (j < 4 ? 8 : 16) : (j < 4 ? 0 : 8);
    int k    = 2 * j + p + base;
    int a    = t * 16 + n;
    int h    = kc * 32 + k;
    wswz[i] = (__bf16)Wh[(size_t)a * HH + h];
}

// ---------------------------------------------------------------------------
// Kernel 1: bias_ctx[b][a] = Wh_b[a] + W0[a,:]*x0[b,:] + W1*x1 + W2*x2
// ---------------------------------------------------------------------------
__global__ void k_bias_ctx(const float* __restrict__ x0, const float* __restrict__ x1,
                           const float* __restrict__ x2, const float* __restrict__ W0,
                           const float* __restrict__ W1, const float* __restrict__ W2,
                           const float* __restrict__ Whb, float* __restrict__ bias) {
    __shared__ float lx[XX0 + XX1 + XX2];
    int b = blockIdx.x, tid = threadIdx.x;
    for (int i = tid; i < XX0; i += 256) lx[i]             = x0[b * XX0 + i];
    for (int i = tid; i < XX1; i += 256) lx[XX0 + i]       = x1[b * XX1 + i];
    for (int i = tid; i < XX2; i += 256) lx[XX0 + XX1 + i] = x2[b * XX2 + i];
    __syncthreads();
    for (int a = tid; a < AA; a += 256) {
        float acc = Whb[a];
        const float* w0 = &W0[(size_t)a * XX0];
        const float* w1 = &W1[(size_t)a * XX1];
        const float* w2 = &W2[(size_t)a * XX2];
        #pragma unroll 4
        for (int k = 0; k < XX0; ++k) acc += w0[k] * lx[k];
        #pragma unroll 4
        for (int k = 0; k < XX1; ++k) acc += w1[k] * lx[XX0 + k];
        #pragma unroll 4
        for (int k = 0; k < XX2; ++k) acc += w2[k] * lx[XX0 + XX1 + k];
        bias[b * AA + a] = acc;
    }
}

// ---------------------------------------------------------------------------
// Kernel 2: scores[b,s] = sum_a V[a]*relu( hidden[b,s,:]@Wh[a,:] + bias[b,a] )
// WG = 256 thr (8 waves) owns one 16-row M-tile. Hidden tile (16x1024 f32,
// 64 KB) is DMA'd into LDS by the Tensor Data Mover, then converted in place
// to the padded bf16 layout (bf16 region = low 33 KB of the same buffer; the
// chunked register-buffered conversion writes strictly below all future
// reads). Waves split N (4 A-tiles each); K loop = 32 x
// v_wmma_f32_16x16x32_bf16 with A from LDS (two 16B ds loads matching the
// ISA 7.12.2 fragment layout) and B as 32 contiguous bytes/lane from the
// pre-swizzled weights.
// ---------------------------------------------------------------------------
__global__ void k_scores(const float* __restrict__ hs, const __bf16* __restrict__ wswz,
                         const float* __restrict__ bias, const float* __restrict__ Vw,
                         float* __restrict__ scoresOut) {
    __shared__ __align__(32) float ldsBufF[16 * 1024];          // 64 KB staging buffer
    __bf16* ldsH     = (__bf16*)ldsBufF;                        // [16][HPAD] bf16 (33 KB)
    float*  stage    = ldsBufF;                                 // 16x1024 f32 TDM tile
    float*  scoreLDS = (float*)((char*)ldsBufF + 40960);        // dead zone after convert

    int tid    = threadIdx.x;
    int wgRow0 = blockIdx.x * 16;       // first flattened (b*S+s) row
    int b      = wgRow0 >> 11;          // S = 2048

#if HAVE_TDM
    if (tid < 32) {                     // wave 0 issues the tensor DMA
        uint64_t ga     = (uint64_t)(uintptr_t)(hs + (size_t)wgRow0 * HH);
        unsigned ldsOff = (unsigned)(uintptr_t)stage;           // low 32b = LDS offset
        // D# group0: count=1 | lds_addr | global_addr[56:0] | type=2
        v4u g0 = { 1u, ldsOff, (unsigned)ga,
                   (unsigned)((ga >> 32) & 0x01FFFFFFu) | (2u << 30) };
        // D# group1: data_size=4B; tensor_dim0=1024, tensor_dim1=16;
        // tile_dim0=1024, tile_dim1=16; tensor_dim0_stride=1024
        v8i g1 = { (int)(2u << 16),        // data_size=2 (4 bytes), mask=0
                   (int)(1024u << 16),     // tensor_dim0[15:0]
                   (int)(16u << 16),       // tensor_dim0[31:16]=0 | tensor_dim1[15:0]=16
                   (int)(1024u << 16),     // tensor_dim1[31:16]=0 | tile_dim0=1024
                   16,                     // tile_dim1=16, tile_dim2=0
                   1024,                   // tensor_dim0_stride[31:0]
                   0, 0 };
        v4i z4 = { 0, 0, 0, 0 };
#if __clang_major__ >= 23
        v8i z8 = { 0, 0, 0, 0, 0, 0, 0, 0 };
        __builtin_amdgcn_tensor_load_to_lds(g0, g1, z4, z4, z8, 0);
#else
        __builtin_amdgcn_tensor_load_to_lds(g0, g1, z4, z4, 0);
#endif
        __builtin_amdgcn_s_wait_tensorcnt(0);
    }
    __syncthreads();
    // In-place f32 -> padded-bf16 conversion, 8 chunks of 2048 elements.
    // Chunk c: reads f32 bytes [c*8192, c*8192+8192), writes bf16 bytes
    // < 33 KB and always below the read frontier of later chunks.
    for (int c = 0; c < 8; ++c) {
        int i0 = c * 2048 + tid * 8;
        float r[8];
        #pragma unroll
        for (int j = 0; j < 8; ++j) r[j] = stage[i0 + j];
        __syncthreads();
        int rr = i0 >> 10, h0 = i0 & 1023;
        #pragma unroll
        for (int j = 0; j < 8; ++j) ldsH[rr * HPAD + h0 + j] = (__bf16)r[j];
        __syncthreads();
    }
#else
    // Fallback: manual global->bf16->LDS staging
    for (int i = tid; i < 16 * 1024; i += 256) {
        int r = i >> 10, h = i & 1023;
        ldsH[r * HPAD + h] = (__bf16)hs[(size_t)wgRow0 * HH + i];
    }
#endif
    if (tid < 16) scoreLDS[tid] = 0.0f;
    __syncthreads();

    int lane = tid & 31;
    int wv   = tid >> 5;                // wave id 0..7 -> owns A-tiles wv, wv+8, wv+16, wv+24
    int hi   = lane >> 4;
    int q    = lane & 15;               // both A-row (M) and B-col (N) index for this lane

    float sAcc[8];
    #pragma unroll
    for (int v = 0; v < 8; ++v) sAcc[v] = 0.0f;

    const __bf16* arow = ldsH + q * HPAD + hi * 8;

    for (int i4 = 0; i4 < 4; ++i4) {
        int t       = wv + (i4 << 3);
        float biasN = bias[b * AA + t * 16 + q];
        float vN    = Vw[t * 16 + q];
        v8f acc = {};
        const v16bf* wtile = (const v16bf*)(wswz + (size_t)(t * 32) * 512);
        #pragma unroll 4
        for (int kc = 0; kc < 32; ++kc) {
            v8bf a0 = *(const v8bf*)(arow + kc * 32);
            v8bf a1 = *(const v8bf*)(arow + kc * 32 + 16);
            v16bf af = __builtin_shufflevector(a0, a1, 0, 1, 2, 3, 4, 5, 6, 7,
                                                       8, 9, 10, 11, 12, 13, 14, 15);
            v16bf bfrag = wtile[kc * 32 + lane];
            acc = __builtin_amdgcn_wmma_f32_16x16x32_bf16(
                false, af, false, bfrag, (short)0, acc, false, false);
        }
        // C layout: lane half hi holds rows hi*8+v (v=0..7), col N=q.
        // relu+scale, then butterfly-sum over the 16 N-lanes of this half.
        #pragma unroll
        for (int v = 0; v < 8; ++v) {
            float c = acc[v] + biasN;
            c = fmaxf(c, 0.0f) * vN;
            c += __shfl_xor(c, 1, 32);
            c += __shfl_xor(c, 2, 32);
            c += __shfl_xor(c, 4, 32);
            c += __shfl_xor(c, 8, 32);
            sAcc[v] += c;
        }
    }

    if (q == 0) {   // one lane per half contributes its 8 row-sums
        #pragma unroll
        for (int v = 0; v < 8; ++v) atomicAdd(&scoreLDS[hi * 8 + v], sAcc[v]);
    }
    __syncthreads();
    if (tid < 16) scoresOut[wgRow0 + tid] = scoreLDS[tid];
}

// ---------------------------------------------------------------------------
// Kernel 3: length-masked softmax over S per batch row -> attention_prob
// ---------------------------------------------------------------------------
__global__ void k_softmax(const float* __restrict__ scores, const int* __restrict__ lengths,
                          float* __restrict__ prob) {
    __shared__ float red[256];
    int b = blockIdx.x, tid = threadIdx.x;
    int len = lengths[b];
    float v[8];
    float m = -INFINITY;
    #pragma unroll
    for (int j = 0; j < 8; ++j) {
        int s = tid + j * 256;
        float sc = scores[b * SS + s];
        v[j] = (s < len) ? sc : -INFINITY;
        m = fmaxf(m, v[j]);
    }
    red[tid] = m;
    __syncthreads();
    for (int o = 128; o > 0; o >>= 1) {
        if (tid < o) red[tid] = fmaxf(red[tid], red[tid + o]);
        __syncthreads();
    }
    m = red[0];
    __syncthreads();
    float lsum = 0.0f;
    #pragma unroll
    for (int j = 0; j < 8; ++j) {
        v[j] = __expf(v[j] - m);   // exp(-inf) = 0 for masked slots
        lsum += v[j];
    }
    red[tid] = lsum;
    __syncthreads();
    for (int o = 128; o > 0; o >>= 1) {
        if (tid < o) red[tid] += red[tid + o];
        __syncthreads();
    }
    float inv = 1.0f / red[0];
    #pragma unroll
    for (int j = 0; j < 8; ++j) prob[b * SS + tid + j * 256] = v[j] * inv;
}

// ---------------------------------------------------------------------------
// Kernel 4: output[b,h] = sum_s prob[b,s] * hidden[b,s,h]
// block = (b, 256-wide h chunk); prob staged in LDS; coalesced hidden stream
// with global_prefetch_b8 running ~128 rows ahead.
// ---------------------------------------------------------------------------
__global__ void k_weighted(const float* __restrict__ hs, const float* __restrict__ prob,
                           float* __restrict__ out) {
    __shared__ float lp[SS];
    int b  = blockIdx.x >> 2;
    int hc = blockIdx.x & 3;
    int tid = threadIdx.x;
    for (int i = tid; i < SS; i += 256) lp[i] = prob[b * SS + i];
    __syncthreads();
    int h = hc * 256 + tid;
    const float* hb = hs + (size_t)b * SS * HH + h;
    float acc = 0.0f;
    for (int s = 0; s < SS; s += 16) {
        int sp = s + 128;
        if (sp < SS) __builtin_prefetch(&hb[(size_t)sp * HH], 0, 0);
        #pragma unroll
        for (int j = 0; j < 16; ++j) acc += lp[s + j] * hb[(size_t)(s + j) * HH];
    }
    out[b * HH + h] = acc;
}

// ---------------------------------------------------------------------------
extern "C" void kernel_launch(void* const* d_in, const int* in_sizes, int n_in,
                              void* d_out, int out_size, void* d_ws, size_t ws_size,
                              hipStream_t stream) {
    (void)in_sizes; (void)n_in; (void)out_size; (void)ws_size;
    const float* hs   = (const float*)d_in[0];
    const float* x0   = (const float*)d_in[1];
    const float* x1   = (const float*)d_in[2];
    const float* x2   = (const float*)d_in[3];
    const int*   len  = (const int*)d_in[4];
    const float* Wh_w = (const float*)d_in[5];
    const float* Wh_b = (const float*)d_in[6];
    const float* W0_w = (const float*)d_in[7];
    const float* W1_w = (const float*)d_in[8];
    const float* W2_w = (const float*)d_in[9];
    const float* V_w  = (const float*)d_in[10];

    float* out     = (float*)d_out;           // [B*H] output, then [B*S] prob
    float* probOut = out + BB * HH;

    char*   ws     = (char*)d_ws;
    __bf16* wswz   = (__bf16*)ws;                               // 1 MiB swizzled bf16 Wh
    float*  bias   = (float*)(ws + (1 << 20));                  // 128 KB bias_ctx
    float*  scores = (float*)(ws + (1 << 20) + (128 << 10));    // 512 KB scores

    k_swizzle_wh<<<(32 * 32 * 512) / 256, 256, 0, stream>>>(Wh_w, wswz);
    k_bias_ctx<<<BB, 256, 0, stream>>>(x0, x1, x2, W0_w, W1_w, W2_w, Wh_b, bias);
    k_scores<<<(BB * SS) / 16, 256, 0, stream>>>(hs, wswz, bias, V_w, scores);
    k_softmax<<<BB, 256, 0, stream>>>(scores, len, probOut);
    k_weighted<<<BB * 4, 256, 0, stream>>>(hs, probOut, out);
}
